// BiLSTM_Each_Cluster_6098853560950
// MI455X (gfx1250) — matrix operand order repaired and checked by reading, not compiled, not run on hardware
//
#include <hip/hip_runtime.h>
#include <cstdint>
#include <cstddef>

// ---------------------------------------------------------------------------
// Types for CDNA5 WMMA (wave32, v_wmma_f32_16x16x32_bf16)
// ---------------------------------------------------------------------------
typedef __bf16 bf16;
typedef __attribute__((ext_vector_type(16))) __bf16 v16bf;
typedef __attribute__((ext_vector_type(8)))  __bf16 v8bf;
typedef __attribute__((ext_vector_type(8)))  float  v8f;

#define B_  64
#define T_  1024
#define D_  256
#define H_  256
#define G4_ 1024        // 4*H
#define O_  64
#define MTB (T_ * B_)   // 65536 token rows

// --------------------------- async-LDS feature probe -----------------------
#if defined(__has_builtin)
#if __has_builtin(__builtin_amdgcn_global_load_async_to_lds_b128) && \
    __has_builtin(__builtin_amdgcn_s_wait_asynccnt)
#define HAVE_ASYNC_LDS 1
#endif
#endif

#if defined(HAVE_ASYNC_LDS)
// Exact parameter type per hipcc diagnostic:
//   '__attribute__((__vector_size__(4 * sizeof(int)))) int __device__ *'
// i.e. int4* in address space 1 (global); LDS side is int4* in AS 3.
typedef int v4i_ __attribute__((vector_size(4 * sizeof(int))));
typedef __attribute__((address_space(1))) v4i_ gv4i;
typedef __attribute__((address_space(3))) v4i_ sv4i;
#endif

union BFrag { v16bf v; v8bf h[2]; };

__device__ __forceinline__ v8f wmma_bf16(v16bf a, v16bf b, v8f c) {
  // 8 args: (neg_a, A, neg_b, B, c_mod, C, reuse_a, reuse_b)
  return __builtin_amdgcn_wmma_f32_16x16x32_bf16(false, a, false, b, (short)0, c,
                                                 false, false);
}

// A-matrix 16x32 bf16 fragment, row-major source, leading dim ld.
// lanes 0-15 -> M=lane, kb=0 ; lanes 16-31 -> M=lane-16, kb=8.
__device__ __forceinline__ v16bf load_a_frag(const bf16* tile, int ld, int lane) {
  BFrag u;
  const int m  = lane & 15;
  const int kb = (lane < 16) ? 0 : 8;
  const bf16* p = tile + (size_t)m * ld + kb;
  u.h[0] = *(const v8bf*)(p);        // K = kb .. kb+7
  u.h[1] = *(const v8bf*)(p + 16);   // K = 16+kb .. 16+kb+7
  return u.v;
}

// B-matrix 32x16 bf16 fragment from row-major W[N,K]:
// lanes 0-15 -> N=lane, K=0..7,16..23 ; lanes 16-31 -> N=lane-16, K=8..15,24..31.
__device__ __forceinline__ v16bf load_b_frag(const bf16* wtile, int ld, int lane) {
  BFrag u;
  const int n  = lane & 15;
  const int kb = (lane < 16) ? 0 : 8;
  const bf16* p = wtile + (size_t)n * ld + kb;
  u.h[0] = *(const v8bf*)(p);
  u.h[1] = *(const v8bf*)(p + 16);
  return u.v;
}

__device__ __forceinline__ float sigmoidf_(float x) {
  return 1.0f / (1.0f + __expf(-x));
}

// Stage one 64-col x 32-k bf16 weight chunk (4KB) into LDS; 16B per thread.
__device__ __forceinline__ void stage_w_chunk(const bf16* __restrict__ W, int K,
                                              int ncol0, int k0, bf16* dst,
                                              int tid) {
  const int r = tid >> 2;
  const int part = (tid & 3) * 8;
  const bf16* g = W + (size_t)(ncol0 + r) * K + k0 + part;
  bf16* l = dst + r * 32 + part;
#if defined(HAVE_ASYNC_LDS)
  __builtin_amdgcn_global_load_async_to_lds_b128(
      (gv4i*)(uintptr_t)g, (sv4i*)(uint32_t)(uintptr_t)l, 0, 0);
#else
  *(v8bf*)l = *(const v8bf*)g;
#endif
}

__device__ __forceinline__ void wait_stage(bool more_inflight) {
#if defined(HAVE_ASYNC_LDS)
  if (more_inflight) __builtin_amdgcn_s_wait_asynccnt(1);
  else               __builtin_amdgcn_s_wait_asynccnt(0);
#else
  (void)more_inflight;
#endif
}

// ---------------------------------------------------------------------------
// Elementwise casts
// ---------------------------------------------------------------------------
__global__ __launch_bounds__(256) void cast_f32_to_bf16(const float* __restrict__ s,
                                                        bf16* __restrict__ d, int n) {
  int i = blockIdx.x * 256 + threadIdx.x;
  if (i < n) d[i] = (bf16)s[i];
}

// x [B,T,D] f32 -> xT [T,B,D] bf16
__global__ __launch_bounds__(256) void transpose_cast_x(const float* __restrict__ x,
                                                        bf16* __restrict__ xT) {
  size_t i = (size_t)blockIdx.x * 256 + threadIdx.x;
  if (i >= (size_t)B_ * T_ * D_) return;
  int d = (int)(i & (D_ - 1));
  size_t r = i >> 8;                 // D_=256
  int t = (int)(r & (T_ - 1));
  int b = (int)(r >> 10);            // T_=1024
  xT[((size_t)t * B_ + b) * D_ + d] = (bf16)x[i];
}

// ---------------------------------------------------------------------------
// Parallel NT GEMM: C[m,n] = sum_k A[m,k]*W[n,k] + bias[n]   (C in bf16)
// Block = 8 waves = 8 m-tiles (128 rows) sharing one 64-column weight panel,
// double-buffer staged through LDS (async-to-LDS when available).
// ---------------------------------------------------------------------------
__global__ __launch_bounds__(256) void gemm_nt_bias_bf16(
    const bf16* __restrict__ A, const bf16* __restrict__ W,
    const float* __restrict__ bias, bf16* __restrict__ C,
    int M, int N, int K) {
  const int tid   = threadIdx.x;
  const int lane  = tid & 31;
  const int wv    = tid >> 5;
  const int nbcnt = N >> 6;
  const int nb    = blockIdx.x % nbcnt;
  const int mb    = blockIdx.x / nbcnt;
  const int mt    = mb * 8 + wv;
  const int ncol0 = nb * 64;
  if (mt * 16 >= M) return;

  __shared__ bf16 wstage[2][64 * 32];      // 2 x 4 KB double buffer

  const int nchunks = K >> 5;
  stage_w_chunk(W, K, ncol0, 0, wstage[0], tid);

  v8f acc[4] = {};
  const bf16* Arow = A + (size_t)mt * 16 * K;

  for (int i = 0; i < nchunks; ++i) {
    if (i + 1 < nchunks)
      stage_w_chunk(W, K, ncol0, (i + 1) * 32, wstage[(i + 1) & 1], tid);
    wait_stage(i + 1 < nchunks);
    __syncthreads();                       // chunk i visible to all waves
    v16bf a = load_a_frag(Arow + i * 32, K, lane);
    const bf16* ws0 = wstage[i & 1];
#pragma unroll
    for (int s = 0; s < 4; ++s)
      acc[s] = wmma_bf16(a, load_b_frag(ws0 + s * 16 * 32, 32, lane), acc[s]);
    __syncthreads();                       // reads done before buffer reuse
  }

  const int laneN  = lane & 15;
  const int rowOff = (lane < 16) ? 0 : 8;
#pragma unroll
  for (int s = 0; s < 4; ++s) {
    int n = ncol0 + s * 16 + laneN;
    float bv = bias[n];
#pragma unroll
    for (int r = 0; r < 8; ++r) {
      size_t m = (size_t)mt * 16 + rowOff + r;
      C[m * N + n] = (bf16)(acc[s][r] + bv);
    }
  }
}

// ---------------------------------------------------------------------------
// Persistent recurrent scan, one block per direction (blockIdx.x: 0=fwd,1=bwd).
// 1024 threads = 32 waves; h in LDS (bf16 64x256), c in registers.
// All 8 A-fragments (h) are hoisted out of the K loop; the 64 (k,tile) WMMAs
// run as a 1-deep software pipeline over the Whh B-fragment loads.
// ---------------------------------------------------------------------------
__global__ __launch_bounds__(1024, 1) void lstm_scan(
    const bf16* __restrict__ gx_f, const bf16* __restrict__ gx_b,
    const bf16* __restrict__ Whh_f, const bf16* __restrict__ Whh_b,
    bf16* __restrict__ out /* [T,B,2H] */) {
  const int dir = blockIdx.x;
  const bf16* __restrict__ gx  = dir ? gx_b : gx_f;
  const bf16* __restrict__ Whh = dir ? Whh_b : Whh_f;

  __shared__ bf16 hbuf[B_ * H_];     // 32 KB of 320 KB WGP LDS

  for (int i = threadIdx.x; i < B_ * H_; i += 1024) hbuf[i] = (bf16)0.0f;

  const int lane   = threadIdx.x & 31;
  const int wv     = threadIdx.x >> 5;          // 0..31
  const int idx0   = wv * 2;                    // 2 of 64 tile positions
  const int mt     = idx0 >> 4;                 // 0..3   (batch tile)
  const int nt0    = idx0 & 15;                 // 0..14  (hidden tile, even)
  const int laneN  = lane & 15;
  const int rowOff = (lane < 16) ? 0 : 8;

  float cst[2][8];
#pragma unroll
  for (int p = 0; p < 2; ++p)
#pragma unroll
    for (int r = 0; r < 8; ++r) cst[p][r] = 0.0f;

  __syncthreads();

  for (int s = 0; s < T_; ++s) {
    const int t = dir ? (T_ - 1 - s) : s;

    // Hoist h(t-1) A-fragments out of the MMA chain (LDS -> 64 VGPRs).
    v16bf afr[8];
#pragma unroll
    for (int kc = 0; kc < 8; ++kc)
      afr[kc] = load_a_frag(hbuf + (size_t)mt * 16 * H_ + kc * 32, H_, lane);

    v8f acc[2][4] = {};
    // q = kc*8 + (p*4+g): 64 WMMA, B-frag loads pipelined one tile ahead.
    v16bf bcur = load_b_frag(Whh + (size_t)(0 * H_ + nt0 * 16) * H_ + 0, H_, lane);
#pragma unroll
    for (int q = 0; q < 64; ++q) {
      const int kc = q >> 3, pg = q & 7, p = pg >> 2, g = pg & 3;
      v16bf bnext = bcur;
      if (q < 63) {
        const int q1 = q + 1;
        const int kc1 = q1 >> 3, pg1 = q1 & 7, p1 = pg1 >> 2, g1 = pg1 & 3;
        bnext = load_b_frag(
            Whh + (size_t)(g1 * H_ + (nt0 + p1) * 16) * H_ + kc1 * 32, H_, lane);
      }
      acc[p][g] = wmma_bf16(afr[kc], bcur, acc[p][g]);
      bcur = bnext;
    }

    __syncthreads();   // all waves done reading h(t-1) from LDS

    // hint the next step's gx slice toward the caches
    if (lane == 0 && s + 1 < T_) {
      int tn = dir ? (T_ - 2 - s) : (s + 1);
      __builtin_prefetch(gx + (size_t)tn * B_ * G4_, 0, 1);
    }

#pragma unroll
    for (int p = 0; p < 2; ++p) {
      const int j = (nt0 + p) * 16 + laneN;     // hidden index 0..255
#pragma unroll
      for (int r = 0; r < 8; ++r) {
        const int b = mt * 16 + rowOff + r;
        const bf16* gp = gx + ((size_t)t * B_ + b) * G4_;
        float gi = acc[p][0][r] + (float)gp[0 * H_ + j];
        float gf = acc[p][1][r] + (float)gp[1 * H_ + j];
        float gg = acc[p][2][r] + (float)gp[2 * H_ + j];
        float go = acc[p][3][r] + (float)gp[3 * H_ + j];
        float c  = sigmoidf_(gf) * cst[p][r] + sigmoidf_(gi) * tanhf(gg);
        cst[p][r] = c;
        float hv = sigmoidf_(go) * tanhf(c);
        bf16 hb = (bf16)hv;
        hbuf[(size_t)b * H_ + j] = hb;
        out[((size_t)t * B_ + b) * (2 * H_) + dir * H_ + j] = hb;
      }
    }

    __syncthreads();   // h(t) visible before next step's A fragments
  }
}

// ---------------------------------------------------------------------------
// FC + exp: out[b,t,o] = exp( h2[t,b,:512] . fcW[o,:512] + fcb[o] ), f32 out.
// Same LDS-staged weight panel scheme (N=64 -> single panel per block).
// ---------------------------------------------------------------------------
__global__ __launch_bounds__(256) void fc_exp_kernel(
    const bf16* __restrict__ A /* [T*B, 512] */, const bf16* __restrict__ W,
    const float* __restrict__ bias, float* __restrict__ out /* [B,T,64] */) {
  const int tid  = threadIdx.x;
  const int lane = tid & 31;
  const int mt   = blockIdx.x * 8 + (tid >> 5);
  if (mt * 16 >= MTB) return;

  __shared__ bf16 wstage[2][64 * 32];

  const int K = 2 * H_;
  const int nchunks = K >> 5;      // 16
  stage_w_chunk(W, K, 0, 0, wstage[0], tid);

  v8f acc[4] = {};
  const bf16* Arow = A + (size_t)mt * 16 * K;

  for (int i = 0; i < nchunks; ++i) {
    if (i + 1 < nchunks)
      stage_w_chunk(W, K, 0, (i + 1) * 32, wstage[(i + 1) & 1], tid);
    wait_stage(i + 1 < nchunks);
    __syncthreads();
    v16bf a = load_a_frag(Arow + i * 32, K, lane);
    const bf16* ws0 = wstage[i & 1];
#pragma unroll
    for (int s = 0; s < 4; ++s)
      acc[s] = wmma_bf16(a, load_b_frag(ws0 + s * 16 * 32, 32, lane), acc[s]);
    __syncthreads();
  }

  const int laneN  = lane & 15;
  const int rowOff = (lane < 16) ? 0 : 8;
#pragma unroll
  for (int s = 0; s < 4; ++s) {
    int n = s * 16 + laneN;
    float bv = bias[n];
#pragma unroll
    for (int r = 0; r < 8; ++r) {
      int m = mt * 16 + rowOff + r;       // m = t*64 + b
      int t = m >> 6, b = m & 63;
      out[(((size_t)b * T_) + t) * O_ + n] = __expf(acc[s][r] + bv);
    }
  }
}

// ---------------------------------------------------------------------------
// Host-side orchestration
// ---------------------------------------------------------------------------
extern "C" void kernel_launch(void* const* d_in, const int* in_sizes, int n_in,
                              void* d_out, int out_size, void* d_ws, size_t ws_size,
                              hipStream_t stream) {
  (void)in_sizes; (void)n_in; (void)out_size; (void)ws_size;
  const float* x     = (const float*)d_in[0];
  const float* Wih0f = (const float*)d_in[1];
  const float* Whh0f = (const float*)d_in[2];
  const float* b0f   = (const float*)d_in[3];
  const float* Wih0b = (const float*)d_in[4];
  const float* Whh0b = (const float*)d_in[5];
  const float* b0b   = (const float*)d_in[6];
  const float* Wih1f = (const float*)d_in[7];
  const float* Whh1f = (const float*)d_in[8];
  const float* b1f   = (const float*)d_in[9];
  const float* Wih1b = (const float*)d_in[10];
  const float* Whh1b = (const float*)d_in[11];
  const float* b1b   = (const float*)d_in[12];
  const float* fcW   = (const float*)d_in[13];
  const float* fcb   = (const float*)d_in[14];
  float* out = (float*)d_out;

  uint8_t* ws = (uint8_t*)d_ws;
  size_t off = 0;
  auto carve = [&](size_t elems) -> bf16* {
    bf16* p = (bf16*)(ws + off);
    off = (off + elems * sizeof(bf16) + 255) & ~(size_t)255;
    return p;
  };
  bf16* wih0f = carve((size_t)G4_ * D_);
  bf16* whh0f = carve((size_t)G4_ * H_);
  bf16* wih0b = carve((size_t)G4_ * D_);
  bf16* whh0b = carve((size_t)G4_ * H_);
  bf16* wih1f = carve((size_t)G4_ * 2 * H_);
  bf16* whh1f = carve((size_t)G4_ * H_);
  bf16* wih1b = carve((size_t)G4_ * 2 * H_);
  bf16* whh1b = carve((size_t)G4_ * H_);
  bf16* fcw   = carve((size_t)O_ * 2 * H_);
  bf16* xT    = carve((size_t)MTB * D_);
  bf16* h1    = carve((size_t)MTB * 2 * H_);
  bf16* h2    = carve((size_t)MTB * 2 * H_);
  bf16* gxf   = carve((size_t)MTB * G4_);
  bf16* gxb   = carve((size_t)MTB * G4_);

  auto cast = [&](const float* s, bf16* d, int n) {
    cast_f32_to_bf16<<<(n + 255) / 256, 256, 0, stream>>>(s, d, n);
  };
  cast(Wih0f, wih0f, G4_ * D_);
  cast(Whh0f, whh0f, G4_ * H_);
  cast(Wih0b, wih0b, G4_ * D_);
  cast(Whh0b, whh0b, G4_ * H_);
  cast(Wih1f, wih1f, G4_ * 2 * H_);
  cast(Whh1f, whh1f, G4_ * H_);
  cast(Wih1b, wih1b, G4_ * 2 * H_);
  cast(Whh1b, whh1b, G4_ * H_);
  cast(fcW,   fcw,   O_ * 2 * H_);

  {
    size_t n = (size_t)B_ * T_ * D_;
    transpose_cast_x<<<(unsigned)((n + 255) / 256), 256, 0, stream>>>(x, xT);
  }

  auto gemm = [&](const bf16* A, const bf16* W, const float* bias, bf16* C,
                  int M, int N, int K) {
    int blocks = (M / 128) * (N / 64);
    gemm_nt_bias_bf16<<<blocks, 256, 0, stream>>>(A, W, bias, C, M, N, K);
  };

  // Layer 0: input projections (parallel), then the serial scan.
  gemm(xT, wih0f, b0f, gxf, MTB, G4_, D_);
  gemm(xT, wih0b, b0b, gxb, MTB, G4_, D_);
  lstm_scan<<<2, 1024, 0, stream>>>(gxf, gxb, whh0f, whh0b, h1);

  // Layer 1
  gemm(h1, wih1f, b1f, gxf, MTB, G4_, 2 * H_);
  gemm(h1, wih1b, b1b, gxb, MTB, G4_, 2 * H_);
  lstm_scan<<<2, 1024, 0, stream>>>(gxf, gxb, whh1f, whh1b, h2);

  // Head: exp(h2 @ fcW^T + fcb) -> [B,T,O]
  {
    int blocks = MTB / 128;
    fc_exp_kernel<<<blocks, 256, 0, stream>>>(h2, fcw, fcb, out);
  }
}
